// VRNN_41231686041856
// MI455X (gfx1250) — matrix operand order repaired and checked by reading, not compiled, not run on hardware
//
#include <hip/hip_runtime.h>
#include <hip/hip_bf16.h>

// ---------------- problem dims ----------------
#define T_DIM 200
#define B_DIM 256
#define X_DIMc 128
#define H_DIMc 512
#define Z_DIMc 64
#define O_DIMc 256
#define G_DIMc 1536   // 3*H

typedef __bf16 bf16;
typedef __attribute__((ext_vector_type(16))) __bf16 v16bf;
typedef __attribute__((ext_vector_type(8)))  float  v8f;
typedef __attribute__((ext_vector_type(4)))  unsigned int v4u;

union Frag { v4u q[2]; v16bf v; };

__device__ __forceinline__ bf16 f2bf(float x) {
  union { float f; unsigned u; } c; c.f = x;
  unsigned r = (c.u + 0x7FFFu + ((c.u >> 16) & 1u)) >> 16;
  unsigned short s = (unsigned short)r;
  return __builtin_bit_cast(bf16, s);
}

__device__ __forceinline__ float softplus_f(float x) {
  return (x > 20.f) ? x : log1pf(expf(x));
}
__device__ __forceinline__ float sigmoid_f(float x) {
  return 1.f / (1.f + expf(-x));
}

// A fragment (16x32 bf16): lane holds row m; K chunks [8*lh, +8) and [16+8*lh, +8).
__device__ __forceinline__ v16bf load_a_frag(const bf16* A, int lda, int m, int k, int lane) {
  const int c0 = ((lane >> 4) & 1) * 8;
  const bf16* p = A + (size_t)m * lda + k + c0;
  Frag f;
  f.q[0] = *(const v4u*)(p);
  f.q[1] = *(const v4u*)(p + 16);
  return f.v;
}

// B fragment (32x16 bf16) from pre-transposed weight Wt[N][K], ldb = K.
__device__ __forceinline__ v16bf load_b_frag(const bf16* Bt, int ldb, int n, int k, int lane) {
  const bf16* p = Bt + (size_t)n * ldb + k + ((lane >> 4) & 1) * 8;
  Frag f;
  f.q[0] = *(const v4u*)(p);
  f.q[1] = *(const v4u*)(p + 16);
  return f.v;
}

__device__ __forceinline__ v8f wmma_bf16(v16bf a, v16bf b, v8f c) {
  return __builtin_amdgcn_wmma_f32_16x16x32_bf16(false, a, false, b, (short)0, c, false, false);
}

// One contiguous-K GEMM segment with hoisted per-lane base pointers.
// All loop-carried updates are constant pointer bumps (+32 elems), so the
// backend emits immediate-offset global_load_b128 clauses with no in-loop
// address VALU (avoids WMMA->VALU hazard NOPs).
__device__ __forceinline__ void gemm_seg(v8f (&acc)[2][2],
                                         const bf16* A, int lda,
                                         const bf16* Bt, int ldb, int kb, int kLen,
                                         int mBase, int nBase, int lane) {
  const int mr = lane & 15;
  const int lh = (lane >> 4) & 1;
  const bf16* a0p = A + (size_t)(mBase + mr) * lda + lh * 8;
  const bf16* a1p = A + (size_t)(mBase + 16 + mr) * lda + lh * 8;
  const bf16* b0p = Bt + (size_t)(nBase + mr) * ldb + kb + lh * 8;
  const bf16* b1p = Bt + (size_t)(nBase + 16 + mr) * ldb + kb + lh * 8;
  for (int k = 0; k < kLen; k += 32) {
    Frag a0, a1, b0, b1;
    a0.q[0] = *(const v4u*)(a0p);      a0.q[1] = *(const v4u*)(a0p + 16);
    a1.q[0] = *(const v4u*)(a1p);      a1.q[1] = *(const v4u*)(a1p + 16);
    b0.q[0] = *(const v4u*)(b0p);      b0.q[1] = *(const v4u*)(b0p + 16);
    b1.q[0] = *(const v4u*)(b1p);      b1.q[1] = *(const v4u*)(b1p + 16);
    a0p += 32; a1p += 32; b0p += 32; b1p += 32;
    acc[0][0] = wmma_bf16(a0.v, b0.v, acc[0][0]);
    acc[0][1] = wmma_bf16(a0.v, b1.v, acc[0][1]);
    acc[1][0] = wmma_bf16(a1.v, b0.v, acc[1][0]);
    acc[1][1] = wmma_bf16(a1.v, b1.v, acc[1][1]);
  }
}

// ---------------- multi-job GEMM launch ----------------
// Independent GEMMs that share M are packed into one launch; blockIdx.z selects
// the job, blockIdx.y >= ny blocks exit uniformly (EXEC stays all-ones around
// WMMA in live blocks). Job types: GEMM epilogues + an elementwise GRU job.
enum { EP_RELU_BF16 = 0, EP_LIN_F32 = 1, EP_SP_F32 = 2, EP_GRU = 3 };

struct GemmJob {
  const void* a0;    // A segment 0 (bf16)  | GRU: gx (f32)
  const void* a1;    // A segment 1 (bf16)  | GRU: gh (f32)
  const void* bt;    // Wt[N][K] bf16
  const void* bias;  // f32 bias[N]
  void* out;         // output              | GRU: h (f32)
  void* out2;        // unused              | GRU: h_bf (bf16)
  int K0, lda0, lda1, ldb, ldo, K, ny, ep;
};
struct GemmBatch { GemmJob j[3]; };

__global__ void __launch_bounds__(128)
gemm_multi_kernel(GemmBatch batch) {
  const GemmJob jb = batch.j[blockIdx.z];
  if ((int)blockIdx.y >= jb.ny) return;   // uniform per block
  const int tid = threadIdx.x;

  if (jb.ep == EP_GRU) {
    // torch GRU cell, fp32 state; publishes bf16 copy for next step's GEMMs
    const float* gx = (const float*)jb.a0;
    const float* gh = (const float*)jb.a1;
    float* h   = (float*)jb.out;
    bf16*  hbf = (bf16*)jb.out2;
    const int total = B_DIM * H_DIMc;
    const int nth = gridDim.x * jb.ny * 128;
    for (int idx = ((int)blockIdx.x * jb.ny + (int)blockIdx.y) * 128 + tid;
         idx < total; idx += nth) {
      int m = idx >> 9;          // /512
      int n = idx & 511;
      size_t ro = (size_t)m * G_DIMc;
      float r  = sigmoid_f(gx[ro + n] + gh[ro + n]);
      float u  = sigmoid_f(gx[ro + H_DIMc + n] + gh[ro + H_DIMc + n]);
      float nn = tanhf(gx[ro + 2 * H_DIMc + n] + r * gh[ro + 2 * H_DIMc + n]);
      float hv = (1.f - u) * nn + u * h[idx];
      h[idx]   = hv;
      hbf[idx] = f2bf(hv);
    }
    return;
  }

  // ---- GEMM: out[M,N] = epilogue(concat(A0,A1)[M,K] @ Wt[N,K]^T + bias) ----
  const int lane = tid & 31;
  const int w    = tid >> 5;
  const int mBase = blockIdx.x * 64 + ((w >> 1) & 1) * 32;
  const int nBase = blockIdx.y * 64 + (w & 1) * 32;
  const int mr = lane & 15;
  const int lh = (lane >> 4) & 1;

  v8f acc[2][2] = {};
  const int kLen0 = (jb.K < jb.K0) ? jb.K : jb.K0;
  gemm_seg(acc, (const bf16*)jb.a0, jb.lda0, (const bf16*)jb.bt, jb.ldb,
           0, kLen0, mBase, nBase, lane);
  if (jb.K > jb.K0) {
    gemm_seg(acc, (const bf16*)jb.a1, jb.lda1, (const bf16*)jb.bt, jb.ldb,
             jb.K0, jb.K - jb.K0, mBase, nBase, lane);
  }

  const float* bias = (const float*)jb.bias;
  const int ep = jb.ep;
#pragma unroll
  for (int mi = 0; mi < 2; ++mi)
#pragma unroll
    for (int ni = 0; ni < 2; ++ni) {
      const int n = nBase + ni * 16 + mr;
      const float bv = bias[n];
#pragma unroll
      for (int e = 0; e < 8; ++e) {
        const int m = mBase + mi * 16 + lh * 8 + e;
        float v = acc[mi][ni][e] + bv;
        if (ep == EP_RELU_BF16) {
          ((bf16*)jb.out)[(size_t)m * jb.ldo + n] = f2bf(v > 0.f ? v : 0.f);
        } else if (ep == EP_LIN_F32) {
          ((float*)jb.out)[(size_t)m * jb.ldo + n] = v;
        } else {
          ((float*)jb.out)[(size_t)m * jb.ldo + n] = softplus_f(v);
        }
      }
    }
}

// ------- fused latent kernel: em/es/z/kld + phi_z GEMM with z staged in LDS ------
__global__ void __launch_bounds__(128)
latent_kernel(const bf16* __restrict__ e2,
              const bf16* __restrict__ emWt, const float* __restrict__ emB,
              const bf16* __restrict__ esWt, const float* __restrict__ esB,
              const float* __restrict__ eps_t,
              const float* __restrict__ pm_t, const float* __restrict__ ps_t,
              float* __restrict__ z_t, float* __restrict__ em_t,
              float* __restrict__ es_t, float* __restrict__ kld_t,
              const bf16* __restrict__ zWt, const float* __restrict__ zB,
              bf16* __restrict__ pz) {
  __shared__ __align__(16) bf16 zl[64 * 64];
  const int lane = threadIdx.x & 31;
  const int w    = threadIdx.x >> 5;
  const int rb   = blockIdx.x * 64;
  const int mr   = lane & 15;
  const int lh   = (lane >> 4) & 1;

  { // em / es GEMMs (both N=64, K=512) + elementwise z, kld
    const int mBase = rb + ((w >> 1) & 1) * 32;
    const int nBase = (w & 1) * 32;
    v8f am[2][2] = {}, as[2][2] = {};
    const bf16* a0p = e2 + (size_t)(mBase + mr) * H_DIMc + lh * 8;
    const bf16* a1p = e2 + (size_t)(mBase + 16 + mr) * H_DIMc + lh * 8;
    const bf16* m0p = emWt + (size_t)(nBase + mr) * H_DIMc + lh * 8;
    const bf16* m1p = emWt + (size_t)(nBase + 16 + mr) * H_DIMc + lh * 8;
    const bf16* s0p = esWt + (size_t)(nBase + mr) * H_DIMc + lh * 8;
    const bf16* s1p = esWt + (size_t)(nBase + 16 + mr) * H_DIMc + lh * 8;
    for (int k = 0; k < H_DIMc; k += 32) {
      Frag a0, a1, bm0, bm1, bs0, bs1;
      a0.q[0]  = *(const v4u*)(a0p);  a0.q[1]  = *(const v4u*)(a0p + 16);
      a1.q[0]  = *(const v4u*)(a1p);  a1.q[1]  = *(const v4u*)(a1p + 16);
      bm0.q[0] = *(const v4u*)(m0p);  bm0.q[1] = *(const v4u*)(m0p + 16);
      bm1.q[0] = *(const v4u*)(m1p);  bm1.q[1] = *(const v4u*)(m1p + 16);
      bs0.q[0] = *(const v4u*)(s0p);  bs0.q[1] = *(const v4u*)(s0p + 16);
      bs1.q[0] = *(const v4u*)(s1p);  bs1.q[1] = *(const v4u*)(s1p + 16);
      a0p += 32; a1p += 32; m0p += 32; m1p += 32; s0p += 32; s1p += 32;
      am[0][0] = wmma_bf16(a0.v, bm0.v, am[0][0]);
      am[0][1] = wmma_bf16(a0.v, bm1.v, am[0][1]);
      am[1][0] = wmma_bf16(a1.v, bm0.v, am[1][0]);
      am[1][1] = wmma_bf16(a1.v, bm1.v, am[1][1]);
      as[0][0] = wmma_bf16(a0.v, bs0.v, as[0][0]);
      as[0][1] = wmma_bf16(a0.v, bs1.v, as[0][1]);
      as[1][0] = wmma_bf16(a1.v, bs0.v, as[1][0]);
      as[1][1] = wmma_bf16(a1.v, bs1.v, as[1][1]);
    }
#pragma unroll
    for (int mi = 0; mi < 2; ++mi)
#pragma unroll
      for (int ni = 0; ni < 2; ++ni) {
        const int n = nBase + ni * 16 + mr;
#pragma unroll
        for (int e = 0; e < 8; ++e) {
          const int m = mBase + mi * 16 + lh * 8 + e;
          const size_t o = (size_t)m * Z_DIMc + n;
          float emv = am[mi][ni][e] + emB[n];
          float esv = softplus_f(as[mi][ni][e] + esB[n]);
          float zv  = eps_t[o] * esv + emv;
          float pmv = pm_t[o], psv = ps_t[o];
          float d   = emv - pmv;
          float kl  = 0.5f * (2.f * logf(psv) - 2.f * logf(esv)
                              + (esv * esv + d * d) / (psv * psv) - 1.f);
          em_t[o] = emv; es_t[o] = esv; z_t[o] = zv; kld_t[o] = kl;
          zl[(m - rb) * 64 + n] = f2bf(zv);
        }
      }
  }
  __syncthreads();
  { // pz[64,512] = relu(z_lds[64,64] @ phi_z_w + b), K=64, A from LDS (ds_load_b128)
    const int mLoc = (w & 1) * 32;
    for (int c0 = (w >> 1) * 32; c0 < H_DIMc; c0 += 64) {
      v8f acc[2][2] = {};
      for (int k = 0; k < Z_DIMc; k += 32) {
        v16bf a0 = load_a_frag(zl, 64, mLoc + mr,      k, lane);
        v16bf a1 = load_a_frag(zl, 64, mLoc + 16 + mr, k, lane);
        v16bf b0 = load_b_frag(zWt, Z_DIMc, c0 + mr,      k, lane);
        v16bf b1 = load_b_frag(zWt, Z_DIMc, c0 + 16 + mr, k, lane);
        acc[0][0] = wmma_bf16(a0, b0, acc[0][0]);
        acc[0][1] = wmma_bf16(a0, b1, acc[0][1]);
        acc[1][0] = wmma_bf16(a1, b0, acc[1][0]);
        acc[1][1] = wmma_bf16(a1, b1, acc[1][1]);
      }
#pragma unroll
      for (int mi = 0; mi < 2; ++mi)
#pragma unroll
        for (int ni = 0; ni < 2; ++ni) {
          const int n = c0 + ni * 16 + mr;
          const float bv = zB[n];
#pragma unroll
          for (int e = 0; e < 8; ++e) {
            const int m = rb + mLoc + mi * 16 + lh * 8 + e;
            float v = acc[mi][ni][e] + bv;
            pz[(size_t)m * H_DIMc + n] = f2bf(v > 0.f ? v : 0.f);
          }
        }
    }
  }
}

// ---------------- small helpers ----------------
__global__ void transpose_w_kernel(const float* __restrict__ W, bf16* __restrict__ Wt,
                                   int I, int O) {
  int idx = blockIdx.x * blockDim.x + threadIdx.x;
  if (idx >= I * O) return;
  int o = idx / I, i = idx - o * I;
  Wt[idx] = f2bf(W[(size_t)i * O + o]);
}

__global__ void f32_to_bf16_kernel(const float* __restrict__ s, bf16* __restrict__ d, int n) {
  int idx = blockIdx.x * blockDim.x + threadIdx.x;
  if (idx < n) d[idx] = f2bf(s[idx]);
}

__global__ void init_h_kernel(float* __restrict__ h, bf16* __restrict__ hbf) {
  int idx = blockIdx.x * blockDim.x + threadIdx.x;
  h[idx] = 0.f;
  hbf[idx] = f2bf(0.f);
}

// ---------------- host side ----------------
extern "C" void kernel_launch(void* const* d_in, const int* in_sizes, int n_in,
                              void* d_out, int out_size, void* d_ws, size_t ws_size,
                              hipStream_t stream) {
  (void)in_sizes; (void)n_in; (void)out_size; (void)ws_size;
  const int T = T_DIM, B = B_DIM, H = H_DIMc, Z = Z_DIMc, O = O_DIMc, G = G_DIMc, X = X_DIMc;
  const size_t TB = (size_t)T * B;

  const float* x        = (const float*)d_in[0];
  const float* eps      = (const float*)d_in[1];
  const float* phi_x_w1 = (const float*)d_in[2];
  const float* phi_x_b1 = (const float*)d_in[3];
  const float* phi_x_w2 = (const float*)d_in[4];
  const float* phi_x_b2 = (const float*)d_in[5];
  const float* phi_z_w  = (const float*)d_in[6];
  const float* phi_z_b  = (const float*)d_in[7];
  const float* enc_w1   = (const float*)d_in[8];
  const float* enc_b1   = (const float*)d_in[9];
  const float* enc_w2   = (const float*)d_in[10];
  const float* enc_b2   = (const float*)d_in[11];
  const float* enc_mw   = (const float*)d_in[12];
  const float* enc_mb   = (const float*)d_in[13];
  const float* enc_sw   = (const float*)d_in[14];
  const float* enc_sb   = (const float*)d_in[15];
  const float* prior_w  = (const float*)d_in[16];
  const float* prior_b  = (const float*)d_in[17];
  const float* prior_mw = (const float*)d_in[18];
  const float* prior_mb = (const float*)d_in[19];
  const float* prior_sw = (const float*)d_in[20];
  const float* prior_sb = (const float*)d_in[21];
  const float* dec_w1   = (const float*)d_in[22];
  const float* dec_b1   = (const float*)d_in[23];
  const float* dec_w2   = (const float*)d_in[24];
  const float* dec_b2   = (const float*)d_in[25];
  const float* dec_lw   = (const float*)d_in[26];
  const float* dec_lb   = (const float*)d_in[27];
  const float* gru_wih  = (const float*)d_in[28];
  const float* gru_whh  = (const float*)d_in[29];
  const float* gru_bih  = (const float*)d_in[30];
  const float* gru_bhh  = (const float*)d_in[31];

  // output slices (scan-stacked, return order)
  float* out   = (float*)d_out;
  float* o_z   = out;
  float* o_log = o_z + TB * Z;
  float* o_em  = o_log + TB * O;
  float* o_es  = o_em + TB * Z;
  float* o_pm  = o_es + TB * Z;
  float* o_ps  = o_pm + TB * Z;
  float* o_kld = o_ps + TB * Z;

  // workspace bump allocator (256B aligned)
  char* base = (char*)d_ws;
  size_t off = 0;
  auto alloc = [&](size_t bytes) -> void* {
    void* p = base + off;
    off += (bytes + 255) & ~(size_t)255;
    return p;
  };

  // transposed bf16 weights [N][K]
  bf16* w_phi1   = (bf16*)alloc((size_t)H * X * 2);
  bf16* w_phi2   = (bf16*)alloc((size_t)H * H * 2);
  bf16* w_phiz   = (bf16*)alloc((size_t)H * Z * 2);
  bf16* w_enc1   = (bf16*)alloc((size_t)H * 2 * H * 2);
  bf16* w_enc2   = (bf16*)alloc((size_t)H * H * 2);
  bf16* w_em     = (bf16*)alloc((size_t)Z * H * 2);
  bf16* w_es     = (bf16*)alloc((size_t)Z * H * 2);
  bf16* w_prior  = (bf16*)alloc((size_t)H * H * 2);
  bf16* w_pm     = (bf16*)alloc((size_t)Z * H * 2);
  bf16* w_ps     = (bf16*)alloc((size_t)Z * H * 2);
  bf16* w_dec1   = (bf16*)alloc((size_t)H * 2 * H * 2);
  bf16* w_dec2   = (bf16*)alloc((size_t)H * H * 2);
  bf16* w_logits = (bf16*)alloc((size_t)O * H * 2);
  bf16* w_gih    = (bf16*)alloc((size_t)G * 2 * H * 2);
  bf16* w_ghh    = (bf16*)alloc((size_t)G * H * 2);
  // activations
  bf16*  x_bf    = (bf16*)alloc(TB * X * 2);
  bf16*  phi1_bf = (bf16*)alloc(TB * H * 2);
  bf16*  px_bf   = (bf16*)alloc(TB * H * 2);
  float* h_f     = (float*)alloc((size_t)B * H * 4);
  bf16*  h_bf    = (bf16*)alloc((size_t)B * H * 2);
  bf16*  pr_bf   = (bf16*)alloc((size_t)B * H * 2);
  bf16*  e1_bf   = (bf16*)alloc((size_t)B * H * 2);
  bf16*  e2_bf   = (bf16*)alloc((size_t)B * H * 2);
  bf16*  pz_bf   = (bf16*)alloc((size_t)B * H * 2);
  bf16*  d1_bf   = (bf16*)alloc((size_t)B * H * 2);
  bf16*  d2_bf   = (bf16*)alloc((size_t)B * H * 2);
  float* gx      = (float*)alloc((size_t)B * G * 4);
  float* gh      = (float*)alloc((size_t)B * G * 4);

  auto tpose = [&](const float* W, bf16* Wt, int I, int Od) {
    int n = I * Od;
    transpose_w_kernel<<<dim3((n + 255) / 256), dim3(256), 0, stream>>>(W, Wt, I, Od);
  };
  auto J = [](const void* a0, const void* a1, int K0, int lda0, int lda1,
              const void* bt, int ldb, const void* bias, void* o, int ldo,
              int K, int ny, int ep) -> GemmJob {
    GemmJob j;
    j.a0 = a0; j.a1 = a1; j.bt = bt; j.bias = bias; j.out = o; j.out2 = nullptr;
    j.K0 = K0; j.lda0 = lda0; j.lda1 = lda1; j.ldb = ldb; j.ldo = ldo;
    j.K = K; j.ny = ny; j.ep = ep;
    return j;
  };
  auto launch1 = [&](int gm, GemmJob a) {
    GemmBatch bt{}; bt.j[0] = a;
    gemm_multi_kernel<<<dim3(gm, a.ny, 1), 128, 0, stream>>>(bt);
  };
  auto launch2 = [&](int gm, GemmJob a, GemmJob b) {
    GemmBatch bt{}; bt.j[0] = a; bt.j[1] = b;
    int gy = a.ny > b.ny ? a.ny : b.ny;
    gemm_multi_kernel<<<dim3(gm, gy, 2), 128, 0, stream>>>(bt);
  };
  auto launch3 = [&](int gm, GemmJob a, GemmJob b, GemmJob c) {
    GemmBatch bt{}; bt.j[0] = a; bt.j[1] = b; bt.j[2] = c;
    int gy = a.ny > b.ny ? a.ny : b.ny; gy = gy > c.ny ? gy : c.ny;
    gemm_multi_kernel<<<dim3(gm, gy, 3), 128, 0, stream>>>(bt);
  };

  // ---- prologue: weight transposes, input convert, h init ----
  tpose(phi_x_w1, w_phi1, X, H);
  tpose(phi_x_w2, w_phi2, H, H);
  tpose(phi_z_w,  w_phiz, Z, H);
  tpose(enc_w1,   w_enc1, 2 * H, H);
  tpose(enc_w2,   w_enc2, H, H);
  tpose(enc_mw,   w_em,   H, Z);
  tpose(enc_sw,   w_es,   H, Z);
  tpose(prior_w,  w_prior, H, H);
  tpose(prior_mw, w_pm,   H, Z);
  tpose(prior_sw, w_ps,   H, Z);
  tpose(dec_w1,   w_dec1, 2 * H, H);
  tpose(dec_w2,   w_dec2, H, H);
  tpose(dec_lw,   w_logits, H, O);
  tpose(gru_wih,  w_gih,  2 * H, G);
  tpose(gru_whh,  w_ghh,  H, G);
  {
    int n = (int)(TB * X);
    f32_to_bf16_kernel<<<dim3((n + 255) / 256), dim3(256), 0, stream>>>(x, x_bf, n);
  }
  init_h_kernel<<<dim3((B * H) / 256), dim3(256), 0, stream>>>(h_f, h_bf);

  // ---- phi_x over all timesteps (two big GEMMs, M = T*B, grid 800x8) ----
  launch1((int)TB / 64, J(x_bf,    x_bf,    X, X, X, w_phi1, X, phi_x_b1, phi1_bf, H, X, H / 64, EP_RELU_BF16));
  launch1((int)TB / 64, J(phi1_bf, phi1_bf, H, H, H, w_phi2, H, phi_x_b2, px_bf,   H, H, H / 64, EP_RELU_BF16));

  // ---- sequential scan: 6 dependent launches per step ----
  const int GM = B / 64;  // 4
  for (int t = 0; t < T; ++t) {
    const bf16*  px_t  = px_bf + (size_t)t * B * H;
    const float* eps_t = eps + (size_t)t * B * Z;
    float* z_t   = o_z   + (size_t)t * B * Z;
    float* log_t = o_log + (size_t)t * B * O;
    float* em_t  = o_em  + (size_t)t * B * Z;
    float* es_t  = o_es  + (size_t)t * B * Z;
    float* pm_t  = o_pm  + (size_t)t * B * Z;
    float* ps_t  = o_ps  + (size_t)t * B * Z;
    float* kld_t = o_kld + (size_t)t * B * Z;

    // L1: prior hidden + enc layer1 (both depend only on h, px)
    launch2(GM,
      J(h_bf, h_bf, H, H, H, w_prior, H, prior_b, pr_bf, H, H, 8, EP_RELU_BF16),
      J(px_t, h_bf, H, H, H, w_enc1, 2 * H, enc_b1, e1_bf, H, 2 * H, 8, EP_RELU_BF16));
    // L2: enc layer2 + prior mean/std heads
    launch3(GM,
      J(e1_bf, e1_bf, H, H, H, w_enc2, H, enc_b2, e2_bf, H, H, 8, EP_RELU_BF16),
      J(pr_bf, pr_bf, H, H, H, w_pm, H, prior_mb, pm_t, Z, H, 1, EP_LIN_F32),
      J(pr_bf, pr_bf, H, H, H, w_ps, H, prior_sb, ps_t, Z, H, 1, EP_SP_F32));
    // L3: fused enc mean/std heads, z sample, kld, phi_z (z via LDS)
    latent_kernel<<<dim3(GM), 128, 0, stream>>>(e2_bf, w_em, enc_mb, w_es, enc_sb,
                                                eps_t, pm_t, ps_t,
                                                z_t, em_t, es_t, kld_t,
                                                w_phiz, phi_z_b, pz_bf);
    // L4: dec layer1 + GRU input/hidden pre-activations (288 blocks)
    launch3(GM,
      J(pz_bf, h_bf,  H, H, H, w_dec1, 2 * H, dec_b1, d1_bf, H, 2 * H, 8, EP_RELU_BF16),
      J(px_t,  pz_bf, H, H, H, w_gih, 2 * H, gru_bih, gx, G, 2 * H, 24, EP_LIN_F32),
      J(h_bf,  h_bf,  H, H, H, w_ghh, H, gru_bhh, gh, G, H, 24, EP_LIN_F32));
    // L5: dec layer2 + GRU state update (GRU depends only on L4)
    {
      GemmJob jgru;
      jgru.a0 = gx; jgru.a1 = gh; jgru.bt = nullptr; jgru.bias = nullptr;
      jgru.out = h_f; jgru.out2 = h_bf;
      jgru.K0 = 0; jgru.lda0 = 0; jgru.lda1 = 0; jgru.ldb = 0; jgru.ldo = 0;
      jgru.K = 0; jgru.ny = 8; jgru.ep = EP_GRU;
      launch2(GM,
        J(d1_bf, d1_bf, H, H, H, w_dec2, H, dec_b2, d2_bf, H, H, 8, EP_RELU_BF16),
        jgru);
    }
    // L6: logits
    launch1(GM, J(d2_bf, d2_bf, H, H, H, w_logits, H, dec_lb, log_t, O, H, 4, EP_LIN_F32));
  }
}
